// SSHENeuralNetworkAggregatorLayer_76441827934768
// MI455X (gfx1250) — matrix operand order repaired and checked by reading, not compiled, not run on hardware
//
#include <hip/hip_runtime.h>

// ---------------------------------------------------------------------------
// z[65536x64] = xa[65536x256] @ wa[256x64] + xb[65536x256] @ wb[256x64]
// Memory-bound streaming GEMM (~144 MB @ 23.3 TB/s ~ 6.2 us floor).
// f32 inputs are split in-register into bf16 hi/lo pairs and multiplied with
// three V_WMMA_F32_16X16X32_BF16 products (xh*wh + xl*wh + xh*wl): near-f32
// accuracy at bf16 matrix throughput -> compute sits far below the HBM roof.
// Weights are staged per-phase into LDS pre-split (hi/lo) and pre-swizzled
// into the exact 32x16 bf16 B-fragment layout (2 conflict-free ds_load_b128
// per fragment).
// ---------------------------------------------------------------------------

typedef __attribute__((ext_vector_type(4)))  float  v4f;
typedef __attribute__((ext_vector_type(8)))  float  v8f;
typedef __attribute__((ext_vector_type(8)))  __bf16 v8bf;
typedef __attribute__((ext_vector_type(16))) __bf16 v16bf;

namespace {
constexpr int kBatch        = 65536;
constexpr int kInK          = 256;               // K per input matrix
constexpr int kOut          = 64;
constexpr int kWaves        = 8;                 // 256 threads, wave32
constexpr int kMTiles       = 2;                 // 16-row tiles per wave
constexpr int kNTiles       = kOut / 16;         // 4
constexpr int kRowsPerWave  = kMTiles * 16;      // 32
constexpr int kRowsPerBlock = kWaves * kRowsPerWave;  // 256
constexpr int kKTile        = 32;                // K per bf16 WMMA
constexpr int kKTilesPhase  = kInK / kKTile;     // 8
// LDS: [h(hi/lo)][ktile][ntile][seg][lane] of v8bf (16B each)
//      = 2*8*4*2*32 = 4096 elements = 64 KB
constexpr int kLdsElems     = 2 * kKTilesPhase * kNTiles * 2 * 32;
}  // namespace

__device__ __forceinline__ int widx(int h, int kt, int nt, int seg, int l) {
  return (((h * kKTilesPhase + kt) * kNTiles + nt) * 2 + seg) * 32 + l;
}

__global__ __launch_bounds__(256)
void sshe_agg_wmma_bf16x3(const float* __restrict__ xa,
                          const float* __restrict__ xb,
                          const float* __restrict__ wa,
                          const float* __restrict__ wb,
                          float* __restrict__ out) {
  __shared__ v8bf wlds[kLdsElems];

  const int tid   = threadIdx.x;
  const int lane  = tid & 31;
  const int wave  = tid >> 5;
  const int lhalf = lane >> 4;   // half-wave select in fragment layouts
  const int lmod  = lane & 15;   // M row (A) / N col (B,C,D)

  const long wave_row0 = (long)blockIdx.x * kRowsPerBlock + (long)wave * kRowsPerWave;

  v8f acc[kMTiles][kNTiles];
#pragma unroll
  for (int m = 0; m < kMTiles; ++m)
#pragma unroll
    for (int n = 0; n < kNTiles; ++n)
      acc[m][n] = (v8f){0.f, 0.f, 0.f, 0.f, 0.f, 0.f, 0.f, 0.f};

#pragma unroll
  for (int phase = 0; phase < 2; ++phase) {
    const float* __restrict__ x = phase ? xb : xa;
    const float* __restrict__ w = phase ? wb : wa;

    // ---- Stage this phase's 256x64 f32 weights into LDS, split hi/lo and
    // pre-swizzled into the B-fragment layout:
    //   lane needs K = lhalf*8 + {0..7} and 16 + lhalf*8 + {0..7} at column
    //   nt*16 + lmod, packed 2 bf16 per dword in K order, 8 dwords per lane
    //   split into two 16B segments (seg 0: first run, seg 1: second run).
    for (int s = tid; s < kLdsElems / 2; s += 256) {
      const int l   = s & 31;
      const int seg = (s >> 5) & 1;
      const int nt  = (s >> 6) & 3;
      const int kt  = s >> 8;
      const int col = nt * 16 + (l & 15);
      const int k0  = kt * kKTile + seg * 16 + (l >> 4) * 8;
      v8bf hi, lo;
#pragma unroll
      for (int j = 0; j < 8; ++j) {
        const float wv = w[(k0 + j) * kOut + col];
        const __bf16 h = (__bf16)wv;
        hi[j] = h;
        lo[j] = (__bf16)(wv - (float)h);
      }
      wlds[widx(0, kt, nt, seg, l)] = hi;
      wlds[widx(1, kt, nt, seg, l)] = lo;
    }
    __syncthreads();

    // ---- Hot loop: 8 K-tiles of 32 ----
#pragma unroll 2
    for (int kt = 0; kt < kKTilesPhase; ++kt) {
      const int kbase = kt * kKTile;

      // A fragments: per 16-bit A 16x32 layout, lane holds two contiguous
      // 8-element K runs: kbase+lhalf*8+{0..7} and kbase+16+lhalf*8+{0..7}.
      v16bf ah[kMTiles], al[kMTiles];
#pragma unroll
      for (int m = 0; m < kMTiles; ++m) {
        const float* p = x + (wave_row0 + m * 16 + lmod) * (long)kInK + kbase;
        const v4f f0 = *(const v4f*)(p + lhalf * 8);
        const v4f f1 = *(const v4f*)(p + lhalf * 8 + 4);
        const v4f f2 = *(const v4f*)(p + 16 + lhalf * 8);
        const v4f f3 = *(const v4f*)(p + 16 + lhalf * 8 + 4);
        float f[16];
#pragma unroll
        for (int i = 0; i < 4; ++i) {
          f[i] = f0[i]; f[4 + i] = f1[i]; f[8 + i] = f2[i]; f[12 + i] = f3[i];
        }
#pragma unroll
        for (int i = 0; i < 16; ++i) {
          const __bf16 h = (__bf16)f[i];
          ah[m][i] = h;
          al[m][i] = (__bf16)(f[i] - (float)h);
        }
      }

#pragma unroll
      for (int n = 0; n < kNTiles; ++n) {
        const v8bf bh0 = wlds[widx(0, kt, n, 0, lane)];
        const v8bf bh1 = wlds[widx(0, kt, n, 1, lane)];
        const v8bf bl0 = wlds[widx(1, kt, n, 0, lane)];
        const v8bf bl1 = wlds[widx(1, kt, n, 1, lane)];
        const v16bf bh = __builtin_shufflevector(
            bh0, bh1, 0, 1, 2, 3, 4, 5, 6, 7, 8, 9, 10, 11, 12, 13, 14, 15);
        const v16bf bl = __builtin_shufflevector(
            bl0, bl1, 0, 1, 2, 3, 4, 5, 6, 7, 8, 9, 10, 11, 12, 13, 14, 15);
#pragma unroll
        for (int m = 0; m < kMTiles; ++m) {
          acc[m][n] = __builtin_amdgcn_wmma_f32_16x16x32_bf16(
              false, ah[m], false, bh, (short)0, acc[m][n], false, false);
          acc[m][n] = __builtin_amdgcn_wmma_f32_16x16x32_bf16(
              false, al[m], false, bh, (short)0, acc[m][n], false, false);
          acc[m][n] = __builtin_amdgcn_wmma_f32_16x16x32_bf16(
              false, ah[m], false, bl, (short)0, acc[m][n], false, false);
        }
      }
    }
    __syncthreads();  // protect wlds before phase-1 restage
  }

  // ---- Store: acc VGPR j holds row (m*16 + j + lhalf*8), col (n*16 + lmod)
#pragma unroll
  for (int m = 0; m < kMTiles; ++m) {
#pragma unroll
    for (int n = 0; n < kNTiles; ++n) {
#pragma unroll
      for (int j = 0; j < 8; ++j) {
        const long row = wave_row0 + m * 16 + j + lhalf * 8;
        out[row * (long)kOut + n * 16 + lmod] = acc[m][n][j];
      }
    }
  }
}

extern "C" void kernel_launch(void* const* d_in, const int* in_sizes, int n_in,
                              void* d_out, int out_size, void* d_ws, size_t ws_size,
                              hipStream_t stream) {
  (void)in_sizes; (void)n_in; (void)out_size; (void)d_ws; (void)ws_size;
  const float* xa = (const float*)d_in[0];
  const float* xb = (const float*)d_in[1];
  const float* wa = (const float*)d_in[2];
  const float* wb = (const float*)d_in[3];
  float* out = (float*)d_out;

  dim3 grid(kBatch / kRowsPerBlock);  // 256 blocks
  dim3 block(256);                    // 8 waves (wave32)
  sshe_agg_wmma_bf16x3<<<grid, block, 0, stream>>>(xa, xb, wa, wb, out);
}
